// MultiHeadAttention_32813550141929
// MI455X (gfx1250) — matrix operand order repaired and checked by reading, not compiled, not run on hardware
//
#include <hip/hip_runtime.h>
#include <hip/hip_bf16.h>

#define DMODEL 1024
#define SEQT   2048
#define NH     16
#define DKH    64
#define BATCH  4
#define MROWS  (BATCH * SEQT)   // 8192

#define LOG2E 1.4426950408889634f

typedef __attribute__((ext_vector_type(16))) __bf16 v16bf;
typedef __attribute__((ext_vector_type(8)))  __bf16 v8bf;
typedef __attribute__((ext_vector_type(8)))  float  v8f;
typedef __attribute__((ext_vector_type(4)))  float  v4f;

static __device__ inline v16bf joinbf(v8bf lo, v8bf hi) {
  v16bf r;
#pragma unroll
  for (int i = 0; i < 8; ++i) { r[i] = lo[i]; r[i + 8] = hi[i]; }
  return r;
}

// load 8 consecutive f32 and convert to bf16 (two b128 loads + pack converts)
static __device__ inline v8bf cvt_v8(const float* p) {
  v4f a = *(const v4f*)p;
  v4f b = *(const v4f*)(p + 4);
  v8bf r;
#pragma unroll
  for (int i = 0; i < 4; ++i) { r[i] = (__bf16)a[i]; r[i + 4] = (__bf16)b[i]; }
  return r;
}

static __device__ inline v8f wmma_bf16(v16bf a, v16bf b, v8f c) {
  return __builtin_amdgcn_wmma_f32_16x16x32_bf16(false, a, false, b, (short)0, c,
                                                 false, false);
}

// DPP16 row_ror:N move (stays within 16-lane rows -> exactly the C-fragment halves)
template <int N>
static __device__ inline float dpp_ror(float v) {
  return __builtin_bit_cast(
      float, __builtin_amdgcn_update_dpp(0, __builtin_bit_cast(int, v),
                                         0x120 | N, 0xf, 0xf, true));
}
static __device__ inline float redmax16(float v) {
  v = fmaxf(v, dpp_ror<1>(v));
  v = fmaxf(v, dpp_ror<2>(v));
  v = fmaxf(v, dpp_ror<4>(v));
  v = fmaxf(v, dpp_ror<8>(v));
  return v;
}
static __device__ inline float redsum16(float v) {
  v += dpp_ror<1>(v);
  v += dpp_ror<2>(v);
  v += dpp_ror<4>(v);
  v += dpp_ror<8>(v);
  return v;
}

// async copy: 16 bytes global -> LDS, tracked by ASYNCcnt (CDNA5 path)
static __device__ inline void async_b128(unsigned ldsAddr, const void* g) {
  asm volatile("global_load_async_to_lds_b128 %0, %1, off"
               :: "v"(ldsAddr), "v"((unsigned long long)(uintptr_t)g)
               : "memory");
}
static __device__ inline void wait_async0() {
  asm volatile("s_wait_asynccnt 0" ::: "memory");
}

// Convert W (K x N, f32, row-major) into Wt (N x K, bf16, row-major).
__global__ __launch_bounds__(256) void wt_convert(const float* __restrict__ W,
                                                  __bf16* __restrict__ Wt) {
  int idx = blockIdx.x * 256 + threadIdx.x;        // over DMODEL*DMODEL
  int n = idx >> 10;
  int k = idx & (DMODEL - 1);
  Wt[idx] = (__bf16)W[(size_t)k * DMODEL + n];
}

// C = (A(M x 1024) * Wt^T + bias) * scale.  Block: 8 waves, tile 128(M) x 64(N).
// B tile (shared by all 8 waves) is staged per K-step into double-buffered LDS
// via GLOBAL_LOAD_ASYNC_TO_LDS_B128: 1 async b128/thread replaces 8 redundant
// global b128/wave.
// OUTMODE 0: bf16 out, (b,h,t,d) per-head layout      (Q, K projections)
// OUTMODE 1: bf16 out, (b,h,d,t) transposed per-head  (V projection)
// OUTMODE 2: f32 out, row-major (M x 1024)            (final projection)
#define BSTRIDE 40   // 32 + 8 pad: 80B rows, 16B-aligned, conflict-free windows
template <typename AT, int OUTMODE>
__global__ __launch_bounds__(256) void gemm_proj(const AT* __restrict__ A,
                                                 const __bf16* __restrict__ Wt,
                                                 const float* __restrict__ bias,
                                                 void* __restrict__ out,
                                                 float scale) {
  const int lane = threadIdx.x & 31;
  const int wave = threadIdx.x >> 5;
  const int l15 = lane & 15;
  const int hi = lane >> 4;
  const int nBlk = blockIdx.x * 64;
  const int mRow = blockIdx.y * 128 + wave * 16 + l15;  // A row for this lane

  __shared__ __attribute__((aligned(16))) __bf16 bT[2][64 * BSTRIDE];
  const int ldN = threadIdx.x >> 2;   // 0..63 : Wt row (N)
  const int ldCh = threadIdx.x & 3;   // 16B chunk within the 64B K-slice
  auto issue_b = [&](int k, int bb) {
    async_b128((unsigned)(uintptr_t)&bT[bb][0] + (unsigned)(ldN * BSTRIDE * 2 + ldCh * 16),
               Wt + (size_t)(nBlk + ldN) * DMODEL + k + ldCh * 8);
  };

  v8f acc[4] = {};
  issue_b(0, 0);   // prologue

#pragma unroll 2
  for (int k = 0; k < DMODEL; k += 32) {
    const int cur = (k >> 5) & 1;
    wait_async0();       // own B-tile loads complete (ASYNCcnt is in-order)
    __syncthreads();     // all waves' loads visible; other buffer now free
    if (k + 32 < DMODEL) issue_b(k + 32, cur ^ 1);  // overlap with compute

    v16bf af;
    if constexpr (sizeof(AT) == 4) {
      const float* ap = (const float*)A + (size_t)mRow * DMODEL + k + 8 * hi;
      af = joinbf(cvt_v8(ap), cvt_v8(ap + 16));
    } else {
      const __bf16* ap = (const __bf16*)A + (size_t)mRow * DMODEL + k + 8 * hi;
      af = joinbf(*(const v8bf*)ap, *(const v8bf*)(ap + 16));
    }
    const __bf16* bTb = &bT[cur][0];
#pragma unroll
    for (int nt = 0; nt < 4; ++nt) {
      const __bf16* bp = bTb + (nt * 16 + l15) * BSTRIDE + 8 * hi;
      v16bf bf = joinbf(*(const v8bf*)bp, *(const v8bf*)(bp + 16));
      acc[nt] = wmma_bf16(af, bf, acc[nt]);
    }
  }

#pragma unroll
  for (int nt = 0; nt < 4; ++nt) {
    const int n = nBlk + nt * 16 + l15;
    const float bv = bias[n];
#pragma unroll
    for (int r = 0; r < 8; ++r) {
      const int row = blockIdx.y * 128 + wave * 16 + r + 8 * hi;
      const float v = (acc[nt][r] + bv) * scale;
      if constexpr (OUTMODE == 2) {
        ((float*)out)[(size_t)row * DMODEL + n] = v;
      } else {
        const int bb = row >> 11, t = row & (SEQT - 1);
        const int h = n >> 6, d = n & (DKH - 1);
        if constexpr (OUTMODE == 0)
          ((__bf16*)out)[(((size_t)(bb * NH + h)) * SEQT + t) * DKH + d] = (__bf16)v;
        else
          ((__bf16*)out)[(((size_t)(bb * NH + h)) * DKH + d) * SEQT + t] = (__bf16)v;
      }
    }
  }
}

// Flash-attention: grid (T/128, B*H), 8 waves/block, each wave = 16 query rows.
// K/V tiles are shared by all 8 waves -> staged once per block into double-
// buffered LDS with GLOBAL_LOAD_ASYNC_TO_LDS_B128 (8x VMEM traffic reduction).
// Q was pre-scaled by 0.125*log2e, so softmax runs in the exp2 domain.
#define PSTRIDE 72   // 64 + 8 pad: 144B rows, 16B-aligned, conflict-free windows
__global__ __launch_bounds__(256) void attn_kernel(const __bf16* __restrict__ Qh,
                                                   const __bf16* __restrict__ Kh,
                                                   const __bf16* __restrict__ Vt,
                                                   __bf16* __restrict__ attnOut) {
  const int lane = threadIdx.x & 31;
  const int wave = threadIdx.x >> 5;
  const int l15 = lane & 15;
  const int hi = lane >> 4;
  const int bh = blockIdx.y;
  const int b = bh >> 4, h = bh & 15;
  const int qBase = blockIdx.x * 128 + wave * 16;

  const __bf16* Qp = Qh + ((size_t)bh * SEQT + qBase) * DKH;
  const __bf16* Kp = Kh + (size_t)bh * SEQT * DKH;
  const __bf16* Vp = Vt + (size_t)bh * DKH * SEQT;

  // K tile: [key 0..63][dk 0..63], V tile: [dk 0..63][key 0..63]
  __shared__ __attribute__((aligned(16))) __bf16 kT[2][64 * PSTRIDE];
  __shared__ __attribute__((aligned(16))) __bf16 vT[2][64 * PSTRIDE];
  __shared__ __attribute__((aligned(16))) __bf16 pT[8][16 * PSTRIDE];

  // cooperative async stage of tile kt into buffer bb: 4x b128 per thread
  const int ldRow = threadIdx.x >> 3;   // 0..31
  const int ldCh = threadIdx.x & 7;     // 16B chunk within a 128B row
  auto issue_tile = [&](int kt, int bb) {
    const int keyBase = kt * 64;
    unsigned kL = (unsigned)(uintptr_t)&kT[bb][0];
    unsigned vL = (unsigned)(uintptr_t)&vT[bb][0];
#pragma unroll
    for (int half = 0; half < 2; ++half) {
      const int r = ldRow + 32 * half;
      const unsigned lofs = (unsigned)(r * PSTRIDE * 2 + ldCh * 16);
      async_b128(kL + lofs, Kp + (size_t)(keyBase + r) * DKH + ldCh * 8);
      async_b128(vL + lofs, Vp + (size_t)r * SEQT + keyBase + ldCh * 8);
    }
  };

  // Resident Q fragments (16 rows x 64 dk = two K-slices)
  v16bf qf[2];
#pragma unroll
  for (int s = 0; s < 2; ++s) {
    const __bf16* qp = Qp + (size_t)l15 * DKH + s * 32 + 8 * hi;
    qf[s] = joinbf(*(const v8bf*)qp, *(const v8bf*)(qp + 16));
  }

  v8f accO[4] = {};
  float mrow[8], lrow[8];
#pragma unroll
  for (int r = 0; r < 8; ++r) { mrow[r] = -1e30f; lrow[r] = 0.f; }

  __bf16* myP = &pT[wave][0];

  issue_tile(0, 0);   // prologue

#pragma unroll 1
  for (int kt = 0; kt < SEQT / 64; ++kt) {
    const int cur = kt & 1;
    wait_async0();       // own tile-kt loads complete (ASYNCcnt is in-order)
    __syncthreads();     // all waves' loads visible; other buffer now free
    if (kt + 1 < SEQT / 64) issue_tile(kt + 1, cur ^ 1);  // overlap with compute

    const __bf16* kTb = &kT[cur][0];
    const __bf16* vTb = &vT[cur][0];

    // S(16x64) = Q * K^T : 8 WMMA, fragments from LDS
    v8f S[4] = {};
#pragma unroll
    for (int ng = 0; ng < 4; ++ng) {
#pragma unroll
      for (int s = 0; s < 2; ++s) {
        const __bf16* kp = kTb + (ng * 16 + l15) * PSTRIDE + s * 32 + 8 * hi;
        v16bf kf = joinbf(*(const v8bf*)kp, *(const v8bf*)(kp + 16));
        S[ng] = wmma_bf16(qf[s], kf, S[ng]);
      }
    }

    // online softmax per query row (C-layout row = r + 8*hi, N = l15)
#pragma unroll
    for (int r = 0; r < 8; ++r) {
      float x0 = S[0][r], x1 = S[1][r], x2 = S[2][r], x3 = S[3][r];
      float mx = redmax16(fmaxf(fmaxf(x0, x1), fmaxf(x2, x3)));
      float mNew = fmaxf(mrow[r], mx);
      float alpha = __builtin_amdgcn_exp2f(mrow[r] - mNew);
      mrow[r] = mNew;
      float p0 = __builtin_amdgcn_exp2f(x0 - mNew);
      float p1 = __builtin_amdgcn_exp2f(x1 - mNew);
      float p2 = __builtin_amdgcn_exp2f(x2 - mNew);
      float p3 = __builtin_amdgcn_exp2f(x3 - mNew);
      lrow[r] = lrow[r] * alpha + redsum16((p0 + p1) + (p2 + p3));
#pragma unroll
      for (int nt = 0; nt < 4; ++nt) accO[nt][r] *= alpha;
      __bf16* pw = myP + (r + 8 * hi) * PSTRIDE + l15;
      pw[0]  = (__bf16)p0;
      pw[16] = (__bf16)p1;
      pw[32] = (__bf16)p2;
      pw[48] = (__bf16)p3;
    }

    // reload P as A-fragments (LDS ops are in-order within a wave)
    v16bf pf[2];
#pragma unroll
    for (int s = 0; s < 2; ++s) {
      const __bf16* pp = myP + l15 * PSTRIDE + s * 32 + 8 * hi;
      pf[s] = joinbf(*(const v8bf*)pp, *(const v8bf*)(pp + 16));
    }

    // O(16x64) += P * V : 8 WMMA, V fragments from LDS (dk-major tile)
#pragma unroll
    for (int s = 0; s < 2; ++s) {
#pragma unroll
      for (int nt = 0; nt < 4; ++nt) {
        const __bf16* vp = vTb + (nt * 16 + l15) * PSTRIDE + s * 32 + 8 * hi;
        v16bf vf = joinbf(*(const v8bf*)vp, *(const v8bf*)(vp + 16));
        accO[nt] = wmma_bf16(pf[s], vf, accO[nt]);
      }
    }
  }

  float inv[8];
#pragma unroll
  for (int r = 0; r < 8; ++r) inv[r] = 1.0f / lrow[r];

  // write to (B, T, H*dk) bf16 so the output projection reads contiguous rows
#pragma unroll
  for (int nt = 0; nt < 4; ++nt) {
    const int col = h * DKH + nt * 16 + l15;
#pragma unroll
    for (int r = 0; r < 8; ++r) {
      const int row = qBase + r + 8 * hi;
      attnOut[((size_t)(b * SEQT + row)) * DMODEL + col] = (__bf16)(accO[nt][r] * inv[r]);
    }
  }
}

extern "C" void kernel_launch(void* const* d_in, const int* in_sizes, int n_in,
                              void* d_out, int out_size, void* d_ws, size_t ws_size,
                              hipStream_t stream) {
  (void)in_sizes; (void)n_in; (void)out_size; (void)ws_size;

  const float* query = (const float*)d_in[0];
  const float* key   = (const float*)d_in[1];
  const float* value = (const float*)d_in[2];
  const float* Wq = (const float*)d_in[3];
  const float* bq = (const float*)d_in[4];
  const float* Wk = (const float*)d_in[5];
  const float* bk = (const float*)d_in[6];
  const float* Wv = (const float*)d_in[7];
  const float* bv = (const float*)d_in[8];
  const float* Wo = (const float*)d_in[9];
  const float* bo = (const float*)d_in[10];

  char* ws = (char*)d_ws;
  __bf16* WtQ = (__bf16*)(ws + (0ull << 20));   // 2 MB each
  __bf16* WtK = (__bf16*)(ws + (2ull << 20));
  __bf16* WtV = (__bf16*)(ws + (4ull << 20));
  __bf16* WtO = (__bf16*)(ws + (6ull << 20));
  __bf16* Qh  = (__bf16*)(ws + (8ull << 20));   // 16 MB each
  __bf16* Kh  = (__bf16*)(ws + (24ull << 20));
  __bf16* Vt  = (__bf16*)(ws + (40ull << 20));
  __bf16* AO  = (__bf16*)(ws + (56ull << 20));  // ends at 72 MB

  dim3 blk(256);

  dim3 gW((DMODEL * DMODEL) / 256);
  wt_convert<<<gW, blk, 0, stream>>>(Wq, WtQ);
  wt_convert<<<gW, blk, 0, stream>>>(Wk, WtK);
  wt_convert<<<gW, blk, 0, stream>>>(Wv, WtV);
  wt_convert<<<gW, blk, 0, stream>>>(Wo, WtO);

  dim3 gG(DMODEL / 64, MROWS / 128);
  // fold the 1/sqrt(dk) score scale AND the log2(e) of the softmax exp into Q
  gemm_proj<float, 0><<<gG, blk, 0, stream>>>(query, WtQ, bq, Qh, 0.125f * LOG2E);
  gemm_proj<float, 0><<<gG, blk, 0, stream>>>(key,   WtK, bk, Kh, 1.0f);
  gemm_proj<float, 1><<<gG, blk, 0, stream>>>(value, WtV, bv, Vt, 1.0f);

  dim3 gA(SEQT / 128, BATCH * NH);
  attn_kernel<<<gA, blk, 0, stream>>>(Qh, Kh, Vt, AO);

  gemm_proj<__bf16, 2><<<gG, blk, 0, stream>>>(AO, WtO, bo, (float*)d_out, 1.0f);
}
